// Generator_5119601017301
// MI455X (gfx1250) — matrix operand order repaired and checked by reading, not compiled, not run on hardware
//
#include <hip/hip_runtime.h>
#include <hip/hip_bf16.h>
#include <math.h>

// Problem dims
#define S_LEN 256
#define B_SZ  64
#define E_DIM 256
#define H_DIM 512
#define H3    1536
#define SB    (S_LEN * B_SZ)

typedef __attribute__((ext_vector_type(16))) __bf16 v16bf;
typedef __attribute__((ext_vector_type(8)))  float  v8f;
typedef __attribute__((ext_vector_type(4)))  int    v4i;
typedef __attribute__((address_space(1))) v4i* gv4i_p;   // global int4*
typedef __attribute__((address_space(3))) v4i* lv4i_p;   // LDS int4*

union V16 {          // 32 bytes: 16 bf16 elements for one WMMA operand
  uint4 q[2];
  v16bf v;
};

__device__ __forceinline__ unsigned short f2bf(float f) {
  unsigned int u = __float_as_uint(f);
  u += 0x7FFFu + ((u >> 16) & 1u);       // round-to-nearest-even
  return (unsigned short)(u >> 16);
}
__device__ __forceinline__ float sigm(float x) { return 1.0f / (1.0f + __expf(-x)); }

// ---------------------------------------------------------------------------
// f32 -> bf16 weight conversion
// ---------------------------------------------------------------------------
__global__ void k_f32_to_bf16(const float* __restrict__ src,
                              unsigned short* __restrict__ dst, int n) {
  int i = blockIdx.x * blockDim.x + threadIdx.x;
  if (i < n) dst[i] = f2bf(src[i]);
}

// ---------------------------------------------------------------------------
// Embedding gather + bf16 convert: embbf[s*B+b][e] = bf16(table[tok][e])
// ---------------------------------------------------------------------------
__global__ void k_embed(const int* __restrict__ sentences,
                        const float* __restrict__ emb_table,
                        unsigned short* __restrict__ embbf) {
  const int sb  = blockIdx.x;                 // s*B + b
  const int tok = sentences[sb];
  const int e   = threadIdx.x;                // blockDim == E_DIM
  embbf[(size_t)sb * E_DIM + e] = f2bf(emb_table[(size_t)tok * E_DIM + e]);
}

// ---------------------------------------------------------------------------
// gi[dir][m][n] = sum_k emb[m][k] * Wih[dir][n][k] + bih[dir][n]
// M=16384, N=1536, K=256.  The 16x256 A panel is staged once into LDS with
// the gfx1250 async-to-LDS copy; all 4 waves read it via ds_load.
// blockDim=128 (4 waves), grid = (M/16, N/256, 2).
// ---------------------------------------------------------------------------
__global__ void __launch_bounds__(128)
k_gemm_gi(const unsigned short* __restrict__ embbf,
          const unsigned short* __restrict__ wihbf,
          const float* __restrict__ bih_f,
          const float* __restrict__ bih_b,
          float* __restrict__ gi) {
  __shared__ unsigned short aLds[16 * E_DIM];           // 8 KB A panel

  const int lane = threadIdx.x & 31;
  const int wave = threadIdx.x >> 5;
  const int col  = lane & 15;
  const int half = lane >> 4;
  const int m0   = blockIdx.x * 16;
  const int nb   = blockIdx.y * 256 + wave * 64;
  const int dir  = blockIdx.z;

  const unsigned short* wih = wihbf + (size_t)dir * H3 * E_DIM;
  const float*          bih = dir ? bih_b : bih_f;
  float*               gout = gi + (size_t)dir * SB * H3;

  // Stage A panel (contiguous 8192 B) global -> LDS.
  {
    const unsigned short* gsrc = embbf + (size_t)m0 * E_DIM;
#if __has_builtin(__builtin_amdgcn_global_load_async_to_lds_b128)
#pragma unroll
    for (int i = 0; i < 4; ++i) {
      const int off = (threadIdx.x + i * 128) * 8;      // 16 B per lane-op
      __builtin_amdgcn_global_load_async_to_lds_b128(
          (gv4i_p)(unsigned short*)(gsrc + off),
          (lv4i_p)(aLds + off), 0, 0);
    }
#if __has_builtin(__builtin_amdgcn_s_wait_asynccnt)
    __builtin_amdgcn_s_wait_asynccnt(0);
#else
    asm volatile("s_wait_asynccnt 0x0" ::: "memory");
#endif
#else
#pragma unroll
    for (int i = 0; i < 4; ++i) {
      const int off = (threadIdx.x + i * 128) * 8;
      *(uint4*)(aLds + off) = *(const uint4*)(gsrc + off);
    }
#endif
  }
  __syncthreads();

  // Per-lane A fragments from LDS (row = lane&15, contiguous K: k = 16*half + s).
  V16 a[8];
  const unsigned short* arow = aLds + col * E_DIM;
#pragma unroll
  for (int kc = 0; kc < 8; ++kc) {
    const unsigned short* p = arow + kc * 32 + half * 16;
    a[kc].q[0] = *(const uint4*)(p);
    a[kc].q[1] = *(const uint4*)(p + 8);
  }

#pragma unroll
  for (int t = 0; t < 4; ++t) {
    const int n0 = nb + t * 16;
    const unsigned short* brow = wih + (size_t)(n0 + col) * E_DIM + half * 16;
    v8f acc = {};
#pragma unroll
    for (int kc = 0; kc < 8; ++kc) {
      V16 b;
      const unsigned short* p = brow + kc * 32;
      b.q[0] = *(const uint4*)(p);
      b.q[1] = *(const uint4*)(p + 8);
      acc = __builtin_amdgcn_wmma_f32_16x16x32_bf16(
          false, a[kc].v, false, b.v, (short)0, acc, false, false);
    }
    const float bias = bih[n0 + col];
#pragma unroll
    for (int r = 0; r < 8; ++r) {
      const int m = m0 + r + half * 8;
      gout[(size_t)m * H3 + n0 + col] = acc[r] + bias;
    }
  }
}

// ---------------------------------------------------------------------------
// Recurrent scan: one 32-wave workgroup per direction; h resident in LDS
// (f32 128KB + bf16 64KB = 192KB of CDNA5's 320KB WGP LDS).
// Each wave owns one j-tile (j0 = wave*16) and all 4 batch-tiles, so the
// three Whh gate panels are loaded once per K-step and reused across
// batch-tiles (2 passes of 2 to bound accumulator pressure), with ping-pong
// double buffering across kc to hide L2 latency under the WMMAs.
// ---------------------------------------------------------------------------
__global__ void __launch_bounds__(1024)
k_recurrent(const unsigned short* __restrict__ whhbf,
            const float* __restrict__ gi,
            const float* __restrict__ bhh_f,
            const float* __restrict__ bhh_b,
            float* __restrict__ hid) {
  extern __shared__ char smem[];
  unsigned short* hbf = (unsigned short*)smem;            // [64][512] bf16
  float*          hf  = (float*)(smem + 65536);           // [64][512] f32

  const int dir = blockIdx.x;
  const unsigned short* whh = whhbf + (size_t)dir * H3 * H_DIM;
  const float* gid = gi  + (size_t)dir * SB * H3;
  const float* bhh = dir ? bhh_b : bhh_f;
  float*       ho  = hid + (size_t)dir * SB * H_DIM;

  for (int i = threadIdx.x; i < B_SZ * H_DIM; i += blockDim.x) {
    hbf[i] = 0;
    hf[i]  = 0.0f;
  }
  __syncthreads();

  const int lane = threadIdx.x & 31;
  const int wave = threadIdx.x >> 5;
  const int col  = lane & 15;
  const int half = lane >> 4;
  const int j0   = wave * 16;          // this wave's hidden-dim tile
  const int jg   = j0 + col;

  // Per-lane Whh row bases for the 3 gates (rows are K-contiguous).
  const unsigned short* br0 = whh + (size_t)(0 * H_DIM + jg) * H_DIM + half * 16;
  const unsigned short* br1 = whh + (size_t)(1 * H_DIM + jg) * H_DIM + half * 16;
  const unsigned short* br2 = whh + (size_t)(2 * H_DIM + jg) * H_DIM + half * 16;
  const float bhr = bhh[jg];
  const float bhz = bhh[H_DIM + jg];
  const float bhn = bhh[2 * H_DIM + jg];

  for (int si = 0; si < S_LEN; ++si) {
    const int s = dir ? (S_LEN - 1 - si) : si;
    float hnew[4][8];

#pragma unroll
    for (int pass = 0; pass < 2; ++pass) {
      const int mbase = pass * 32;
      __builtin_prefetch(gid + (size_t)(s * B_SZ + mbase) * H3 + j0, 0, 1);

      v8f acc[2][3];
#pragma unroll
      for (int mi = 0; mi < 2; ++mi)
#pragma unroll
        for (int g = 0; g < 3; ++g) acc[mi][g] = (v8f){};

      V16 bb[2][3];                       // ping-pong B buffers
      bb[0][0].q[0] = *(const uint4*)(br0); bb[0][0].q[1] = *(const uint4*)(br0 + 8);
      bb[0][1].q[0] = *(const uint4*)(br1); bb[0][1].q[1] = *(const uint4*)(br1 + 8);
      bb[0][2].q[0] = *(const uint4*)(br2); bb[0][2].q[1] = *(const uint4*)(br2 + 8);

#pragma unroll
      for (int kc = 0; kc < 16; ++kc) {   // K = 512
        const int cur = kc & 1;
        const int nxt = cur ^ 1;
        if (kc < 15) {
          const int ko = (kc + 1) * 32;
          bb[nxt][0].q[0] = *(const uint4*)(br0 + ko);
          bb[nxt][0].q[1] = *(const uint4*)(br0 + ko + 8);
          bb[nxt][1].q[0] = *(const uint4*)(br1 + ko);
          bb[nxt][1].q[1] = *(const uint4*)(br1 + ko + 8);
          bb[nxt][2].q[0] = *(const uint4*)(br2 + ko);
          bb[nxt][2].q[1] = *(const uint4*)(br2 + ko + 8);
        }
#pragma unroll
        for (int mi = 0; mi < 2; ++mi) {
          const int m0 = mbase + mi * 16;
          V16 a;
          const unsigned short* ap = hbf + (m0 + col) * H_DIM + kc * 32 + half * 16;
          a.q[0] = *(const uint4*)(ap);
          a.q[1] = *(const uint4*)(ap + 8);
          acc[mi][0] = __builtin_amdgcn_wmma_f32_16x16x32_bf16(
              false, a.v, false, bb[cur][0].v, (short)0, acc[mi][0], false, false);
          acc[mi][1] = __builtin_amdgcn_wmma_f32_16x16x32_bf16(
              false, a.v, false, bb[cur][1].v, (short)0, acc[mi][1], false, false);
          acc[mi][2] = __builtin_amdgcn_wmma_f32_16x16x32_bf16(
              false, a.v, false, bb[cur][2].v, (short)0, acc[mi][2], false, false);
        }
      }

      // Fused GRU gate math for the two batch-tiles of this pass.
#pragma unroll
      for (int mi = 0; mi < 2; ++mi) {
        const int m0 = mbase + mi * 16;
        const int ti = pass * 2 + mi;
#pragma unroll
        for (int r = 0; r < 8; ++r) {
          const int b_i = m0 + r + half * 8;
          const float* grow = gid + (size_t)(s * B_SZ + b_i) * H3;
          const float rr = sigm(grow[jg] + acc[mi][0][r] + bhr);
          const float zg = sigm(grow[H_DIM + jg] + acc[mi][1][r] + bhz);
          const float nn = tanhf(grow[2 * H_DIM + jg] + rr * (acc[mi][2][r] + bhn));
          const float hold = hf[b_i * H_DIM + jg];
          const float hv   = (1.0f - zg) * nn + zg * hold;
          hnew[ti][r] = hv;
          ho[(size_t)(s * B_SZ + b_i) * H_DIM + jg] = hv;
        }
      }
    }

    __syncthreads();   // all waves done reading h for this step
#pragma unroll
    for (int ti = 0; ti < 4; ++ti) {
#pragma unroll
      for (int r = 0; r < 8; ++r) {
        const int b_i = ti * 16 + r + half * 8;
        const float hv = hnew[ti][r];
        hf[b_i * H_DIM + jg]  = hv;
        hbf[b_i * H_DIM + jg] = f2bf(hv);
      }
    }
    __syncthreads();   // h updated for next step
  }
}

// ---------------------------------------------------------------------------
// pz = sigmoid(<hidden, zW> + zb), z = noise < pz.  One wave per (s,b).
// ---------------------------------------------------------------------------
__global__ void k_pz(const float* __restrict__ hid,
                     const float* __restrict__ zW,
                     const float* __restrict__ zb,
                     const float* __restrict__ noise,
                     float* __restrict__ out) {
  const int widx = (blockIdx.x * blockDim.x + threadIdx.x) >> 5;  // s*B+b
  const int lane = threadIdx.x & 31;
  const float* hfp = hid + (size_t)widx * H_DIM;
  const float* hbp = hid + (size_t)SB * H_DIM + (size_t)widx * H_DIM;
  float acc = 0.0f;
  for (int k = lane; k < H_DIM; k += 32)
    acc += hfp[k] * zW[k] + hbp[k] * zW[H_DIM + k];
#pragma unroll
  for (int off = 16; off > 0; off >>= 1)
    acc += __shfl_xor(acc, off, 32);
  if (lane == 0) {
    const float p = sigm(acc + zb[0]);
    out[widx]      = p;                                   // pz
    out[SB + widx] = (noise[widx] < p) ? 1.0f : 0.0f;     // z
  }
}

// ---------------------------------------------------------------------------
// Stable ragged compaction per column b, plus z_sizes.
// ---------------------------------------------------------------------------
__global__ void k_compact(const int* __restrict__ sentences,
                          float* __restrict__ out) {
  __shared__ int sc[S_LEN];
  const int b    = blockIdx.x;
  const int s    = threadIdx.x;
  const int tok  = sentences[s * B_SZ + b];
  const int flag = (out[SB + s * B_SZ + b] > 0.5f) ? 1 : 0;
  sc[s] = flag;
  __syncthreads();
  for (int off = 1; off < S_LEN; off <<= 1) {             // inclusive scan
    int v = sc[s];
    if (s >= off) v += sc[s - off];
    __syncthreads();
    sc[s] = v;
    __syncthreads();
  }
  const int incl  = sc[s];
  const int total = sc[S_LEN - 1];
  float* rat = out + 2 * SB;
  if (flag) rat[(incl - 1) * B_SZ + b] = (float)tok;
  if (s >= total) rat[s * B_SZ + b] = 0.0f;               // PAD_ID
  if (s == 0) out[3 * SB + b] = (float)total;             // z_sizes
}

// ---------------------------------------------------------------------------
extern "C" void kernel_launch(void* const* d_in, const int* in_sizes, int n_in,
                              void* d_out, int out_size, void* d_ws, size_t ws_size,
                              hipStream_t stream) {
  (void)in_sizes; (void)n_in; (void)out_size; (void)ws_size;
  const int*   sentences = (const int*)  d_in[0];
  const float* noise     = (const float*)d_in[1];
  const float* emb_table = (const float*)d_in[2];
  const float* Wih_f = (const float*)d_in[3];
  const float* Whh_f = (const float*)d_in[4];
  const float* bih_f = (const float*)d_in[5];
  const float* bhh_f = (const float*)d_in[6];
  const float* Wih_b = (const float*)d_in[7];
  const float* Whh_b = (const float*)d_in[8];
  const float* bih_b = (const float*)d_in[9];
  const float* bhh_b = (const float*)d_in[10];
  const float* zW    = (const float*)d_in[11];
  const float* zb    = (const float*)d_in[12];

  // Workspace carve-up (bytes):
  //   wihbf  [2][1536][256] bf16 : 1,572,864
  //   whhbf  [2][1536][512] bf16 : 3,145,728
  //   embbf  [16384][256]   bf16 : 8,388,608
  //   gi     [2][16384][1536] f32: 201,326,592
  //   hid    [2][16384][512]  f32: 67,108,864
  char* ws = (char*)d_ws;
  unsigned short* wihbf = (unsigned short*)ws;
  unsigned short* whhbf = (unsigned short*)(ws + 1572864);
  unsigned short* embbf = (unsigned short*)(ws + 4718592);
  float*          gi    = (float*)(ws + 13107200);
  float*          hid   = (float*)(ws + 214433792);

  k_f32_to_bf16<<<(H3 * E_DIM + 255) / 256, 256, 0, stream>>>(Wih_f, wihbf, H3 * E_DIM);
  k_f32_to_bf16<<<(H3 * E_DIM + 255) / 256, 256, 0, stream>>>(Wih_b, wihbf + H3 * E_DIM, H3 * E_DIM);
  k_f32_to_bf16<<<(H3 * H_DIM + 255) / 256, 256, 0, stream>>>(Whh_f, whhbf, H3 * H_DIM);
  k_f32_to_bf16<<<(H3 * H_DIM + 255) / 256, 256, 0, stream>>>(Whh_b, whhbf + H3 * H_DIM, H3 * H_DIM);

  k_embed<<<SB, E_DIM, 0, stream>>>(sentences, emb_table, embbf);

  k_gemm_gi<<<dim3(SB / 16, H3 / 256, 2), 128, 0, stream>>>(
      embbf, wihbf, bih_f, bih_b, gi);

  k_recurrent<<<2, 1024, 196608, stream>>>(whhbf, gi, bhh_f, bhh_b, hid);

  k_pz<<<(SB * 32) / 256, 256, 0, stream>>>(hid, zW, zb, noise, (float*)d_out);

  k_compact<<<B_SZ, S_LEN, 0, stream>>>(sentences, (float*)d_out);
}
